// SelfAttention_26757646254461
// MI455X (gfx1250) — compile-verified, matrix operand order checked by reading, not run on hardware
//
#include <hip/hip_runtime.h>

// ---------------------------------------------------------------------------
// Self-attention on MI455X (gfx1250), compute-bound (~129 GFLOP vs ~70MB HBM
// traffic): all matmuls in bf16 WMMA (v_wmma_f32_16x16x32_bf16, f32 accum).
// GEMM LDS tiles are fed by the Tensor Data Mover (tensor_load_to_lds +
// s_wait_tensorcnt) with hardware row padding, double-buffered so the SIMDs
// only run WMMA + fragment ds_loads. Inputs are pre-converted to bf16 once.
// ---------------------------------------------------------------------------

typedef __attribute__((ext_vector_type(16))) __bf16 v16bf;
typedef __attribute__((ext_vector_type(8)))  __bf16 v8bf;
typedef __attribute__((ext_vector_type(4)))  __bf16 v4bf;
typedef __attribute__((ext_vector_type(8)))  float  v8f;
typedef __attribute__((ext_vector_type(4)))  float  v4f;
typedef __attribute__((ext_vector_type(4)))  unsigned int v4u;
typedef __attribute__((ext_vector_type(4)))  int v4i;
typedef __attribute__((ext_vector_type(8)))  int v8i;

#define TN    2048   // time dim / GEMM N
#define KD    1024   // channel dim / GEMM K (= H*OD)
#define HEADS 8
#define ODIM  128
#define LDPAD 40     // LDS row pitch in bf16: 64B data + 16B pad = 80B

__device__ __forceinline__ v16bf comb(v8bf lo, v8bf hi) {
  return __builtin_shufflevector(lo, hi, 0,1,2,3,4,5,6,7,8,9,10,11,12,13,14,15);
}
__device__ __forceinline__ v8f wmma_bf16(v16bf a, v16bf b, v8f c) {
  return __builtin_amdgcn_wmma_f32_16x16x32_bf16(false, a, false, b, (short)0, c, false, false);
}

// ---------------------------------------------------------------------------
// TDM: DMA one 128-row x 32-col bf16 tile (row pitch `pitch` elements in
// global) into LDS with 80B padded rows. Issued by one wave; tracked by
// TENSORcnt. D# layout per CDNA5 ISA ch.8 (group0: count/lds/global/type,
// group1: data_size=2B, pad every 16 dwords by 4 dwords, dims/strides).
// ---------------------------------------------------------------------------
__device__ __forceinline__ void tdm_load_128x32(__bf16* lds_dst,
                                                const __bf16* gsrc,
                                                unsigned pitch) {
#if __has_builtin(__builtin_amdgcn_tensor_load_to_lds)
  unsigned long long ga = (unsigned long long)(uintptr_t)gsrc;
  unsigned la = (unsigned)(uintptr_t)lds_dst;  // low 32 bits = LDS byte offset
  v4u g0;
  g0.x = 1u;                                   // count=1, user descriptor
  g0.y = la;                                   // lds_addr
  g0.z = (unsigned)ga;                         // global_addr[31:0]
  g0.w = ((unsigned)(ga >> 32) & 0x01FFFFFFu)  // global_addr[56:32]
       | (2u << 30);                           // type = 2 ("image")
  v8i g1;
  g1[0] = (int)((1u << 16)        // data_size = 2 bytes
              | (1u << 20)        // pad_enable
              | (3u << 22)        // pad_interval = 16 dwords (one 64B row)
              | (3u << 25));      // pad_amount   = 4 dwords (16B) -> 80B pitch
  g1[1] = (int)((pitch & 0xFFFFu) << 16);            // tensor_dim0[15:0]
  g1[2] = (int)((pitch >> 16) | (0xFFFFu << 16));    // dim0[31:16] | dim1[15:0]
  g1[3] = (int)(32u << 16);                          // dim1[31:16]=0 | tile_dim0=32
  g1[4] = (int)128u;                                 // tile_dim1=128, tile_dim2=0
  g1[5] = (int)pitch;                                // tensor_dim0_stride[31:0]
  g1[6] = 0;
  g1[7] = 0;
  v4i z4 = {0, 0, 0, 0};
#if __clang_major__ >= 23
  v8i z8 = {0, 0, 0, 0, 0, 0, 0, 0};
  __builtin_amdgcn_tensor_load_to_lds(g0, g1, z4, z4, z8, 0);
#else
  __builtin_amdgcn_tensor_load_to_lds(g0, g1, z4, z4, 0);
#endif
#else
  // Fallback: wave-cooperative copy (probed toolchains all have the builtin).
  int lane = threadIdx.x & 31;
  for (int r = 0; r < 128; ++r) {
    if (lane < 16)
      ((unsigned*)(lds_dst + r * LDPAD))[lane] =
          ((const unsigned*)(gsrc + (size_t)r * pitch))[lane];
  }
#endif
}

// ---------------------------------------------------------------------------
// fp32 -> bf16 bulk convert (weights). n4 = element_count / 4.
// ---------------------------------------------------------------------------
__global__ __launch_bounds__(256) void cvt_bf16(const float* __restrict__ src,
                                                __bf16* __restrict__ dst, int n4) {
  int i = blockIdx.x * 256 + threadIdx.x;
  if (i < n4) {
    v4f s = *(const v4f*)(src + 4 * (size_t)i);
    v4bf d;
    d.x = (__bf16)s.x; d.y = (__bf16)s.y; d.z = (__bf16)s.z; d.w = (__bf16)s.w;
    *(v4bf*)(dst + 4 * (size_t)i) = d;
  }
}

// ---------------------------------------------------------------------------
// x (b, k, t) fp32 -> xT (b, t, k) bf16, 32x32 tiles through LDS.
// ---------------------------------------------------------------------------
__global__ __launch_bounds__(256) void transpose_x(const float* __restrict__ x,
                                                   __bf16* __restrict__ xT) {
  __shared__ __bf16 tile[32][33];
  const int tx = threadIdx.x, ty = threadIdx.y;  // (32, 8)
  const int b = blockIdx.z;
  const int t0 = blockIdx.x * 32, k0 = blockIdx.y * 32;
  const float* xb = x + (size_t)b * KD * TN;
  __bf16* xTb = xT + (size_t)b * TN * KD;
  #pragma unroll
  for (int i = 0; i < 4; ++i)
    tile[ty + 8 * i][tx] = (__bf16)xb[(size_t)(k0 + ty + 8 * i) * TN + t0 + tx];
  __syncthreads();
  #pragma unroll
  for (int i = 0; i < 4; ++i)
    xTb[(size_t)(t0 + ty + 8 * i) * KD + k0 + tx] = tile[tx][ty + 8 * i];
}

// ---------------------------------------------------------------------------
// Batched WMMA GEMM, TDM-fed + double-buffered LDS.
//   C[M x 2048] = A[M x 1024] * B, where Bt holds B^T (n-major: 2048 x 1024)
//   A, Bt bf16; contraction dim contiguous for both -> pure b128 ds fragments.
//   OMODE 0: C -> bf16 (b, j, t)            (V projection)
//   OMODE 1: C -> bf16 (b, h, t, e)         (Q/K projections)
//   OMODE 2: C -> fp32 + bias[j]            (final unify)
// 256 thr = 8 waves; block tile 128x128; wave tile 32x64 (2x4 WMMA tiles).
// ---------------------------------------------------------------------------
template<int OMODE>
__global__ __launch_bounds__(256) void gemm_wmma_tdm(
    const __bf16* __restrict__ A, const __bf16* __restrict__ Bt,
    void* __restrict__ Cout, const float* __restrict__ bias, float scale)
{
  __shared__ __bf16 Alds[2][128 * LDPAD] __attribute__((aligned(16)));
  __shared__ __bf16 Blds[2][128 * LDPAD] __attribute__((aligned(16)));

  const int tid  = threadIdx.x;
  const int lane = tid & 31;
  const int wid  = tid >> 5;
  const int lm   = lane & 15;
  const int h2   = lane >> 4;
  const int wm   = (wid & 3) * 32;   // 4 waves tile M
  const int wn   = (wid >> 2) * 64;  // 2 waves tile N
  const int mblk = blockIdx.y * 128;
  const int nblk = blockIdx.x * 128;
  const int z    = blockIdx.z;

  const __bf16* Ab = A + (size_t)mblk * KD;
  const __bf16* Bb = Bt + (size_t)z * TN * KD + (size_t)nblk * KD;

  v8f zero8 = {};
  v8f acc[2][4];
  #pragma unroll
  for (int mi = 0; mi < 2; ++mi)
    #pragma unroll
    for (int ni = 0; ni < 4; ++ni) acc[mi][ni] = zero8;

  if (wid == 0) {                        // prologue: DMA first tiles
    tdm_load_128x32(&Alds[0][0], Ab, KD);
    tdm_load_128x32(&Blds[0][0], Bb, KD);
  }

  for (int kk = 0; kk < KD; kk += 32) {
    const int cur = (kk >> 5) & 1;
    if (wid == 0) __builtin_amdgcn_s_wait_tensorcnt(0);
    __syncthreads();                     // cur buffers ready for everyone
    if (wid == 0 && kk + 32 < KD) {      // prefetch next tiles into other buf
      tdm_load_128x32(&Alds[cur ^ 1][0], Ab + kk + 32, KD);
      tdm_load_128x32(&Blds[cur ^ 1][0], Bb + kk + 32, KD);
    }

    v16bf af[2];
    #pragma unroll
    for (int mi = 0; mi < 2; ++mi) {
      const __bf16* ap = &Alds[cur][(wm + mi * 16 + lm) * LDPAD];
      af[mi] = comb(*(const v8bf*)(ap + h2 * 8), *(const v8bf*)(ap + 16 + h2 * 8));
    }
    #pragma unroll
    for (int ni = 0; ni < 4; ++ni) {
      const __bf16* bp = &Blds[cur][(wn + ni * 16 + lm) * LDPAD];
      v16bf bfrag = comb(*(const v8bf*)(bp + h2 * 8), *(const v8bf*)(bp + 16 + h2 * 8));
      #pragma unroll
      for (int mi = 0; mi < 2; ++mi)
        acc[mi][ni] = wmma_bf16(af[mi], bfrag, acc[mi][ni]);
    }
    __syncthreads();                     // done reading cur before it is reused
  }

  // epilogue: D layout lane -> (m = r + 8*h2, n = lm)
  #pragma unroll
  for (int mi = 0; mi < 2; ++mi) {
    #pragma unroll
    for (int ni = 0; ni < 4; ++ni) {
      #pragma unroll
      for (int r = 0; r < 8; ++r) {
        int j = mblk + wm + mi * 16 + r + 8 * h2;
        int t = nblk + wn + ni * 16 + lm;
        float v = acc[mi][ni][r] * scale;
        if constexpr (OMODE == 0) {
          ((__bf16*)Cout)[(size_t)z * KD * TN + (size_t)j * TN + t] = (__bf16)v;
        } else if constexpr (OMODE == 1) {
          int h = j >> 7, e = j & 127;
          ((__bf16*)Cout)[(((size_t)z * HEADS + h) * TN + t) * ODIM + e] = (__bf16)v;
        } else {
          ((float*)Cout)[(size_t)z * ODIM * TN + (size_t)j * TN + t] = v + bias[j];
        }
      }
    }
  }
}

// ---------------------------------------------------------------------------
// Flash attention: each wave owns a 16-row t-block of one (b,h), streams keys
// in i-steps of 32 with online softmax.
//   S(16x32)    = Q^T(16x128) * K(128x32)   : 8 WMMAs
//   O^T(16x128) += P(16x32) * V^T(32x128)   : 8 WMMAs
// Q pre-scaled by 1/sqrt(OD). P goes through a per-wave LDS bounce to reach
// the A-fragment layout. Output written as (b, t, j) so the unify GEMM's TDM
// can copy it untransposed.
// ---------------------------------------------------------------------------
__global__ __launch_bounds__(256) void attn_wmma(
    const __bf16* __restrict__ Qt,  // (b,h,t,e), pre-scaled
    const __bf16* __restrict__ Kt,  // (b,h,i,e)
    const __bf16* __restrict__ Vb,  // (b, j=h*128+e, i)
    __bf16* __restrict__ Ot)        // (b, t, j)
{
  __shared__ __bf16 plds[8][16][32] __attribute__((aligned(16)));
  const int tid  = threadIdx.x;
  const int lane = tid & 31;
  const int wid  = tid >> 5;
  const int lm   = lane & 15;
  const int h2   = lane >> 4;
  const int task = blockIdx.x * 8 + wid;
  const int bh   = task >> 7;
  const int tb   = (task & 127) * 16;

  v16bf qf[4];
  {
    const __bf16* qp = Qt + ((size_t)bh * TN + tb + lm) * ODIM;
    #pragma unroll
    for (int es = 0; es < 4; ++es)
      qf[es] = comb(*(const v8bf*)(qp + es * 32 + h2 * 8),
                    *(const v8bf*)(qp + es * 32 + 16 + h2 * 8));
  }

  v8f zero8 = {};
  v8f accO[8];
  #pragma unroll
  for (int nt = 0; nt < 8; ++nt) accO[nt] = zero8;
  float mr[8], lr[8];
  #pragma unroll
  for (int r = 0; r < 8; ++r) { mr[r] = -3.0e38f; lr[r] = 0.0f; }

  for (int i0 = 0; i0 < TN; i0 += 32) {
    v8f s0 = zero8, s1 = zero8;
    const __bf16* kp0 = Kt + ((size_t)bh * TN + i0 + lm) * ODIM;
    const __bf16* kp1 = kp0 + 16 * ODIM;
    #pragma unroll
    for (int es = 0; es < 4; ++es) {
      v16bf k0 = comb(*(const v8bf*)(kp0 + es * 32 + h2 * 8),
                      *(const v8bf*)(kp0 + es * 32 + 16 + h2 * 8));
      v16bf k1 = comb(*(const v8bf*)(kp1 + es * 32 + h2 * 8),
                      *(const v8bf*)(kp1 + es * 32 + 16 + h2 * 8));
      s0 = wmma_bf16(qf[es], k0, s0);
      s1 = wmma_bf16(qf[es], k1, s1);
    }
    float alpha[8];
    #pragma unroll
    for (int r = 0; r < 8; ++r) {
      float mx = fmaxf(s0[r], s1[r]);
      mx = fmaxf(mx, __shfl_xor(mx, 1, 32));
      mx = fmaxf(mx, __shfl_xor(mx, 2, 32));
      mx = fmaxf(mx, __shfl_xor(mx, 4, 32));
      mx = fmaxf(mx, __shfl_xor(mx, 8, 32));
      float mnew = fmaxf(mr[r], mx);
      alpha[r] = __expf(mr[r] - mnew);
      float p0 = __expf(s0[r] - mnew);
      float p1 = __expf(s1[r] - mnew);
      s0[r] = p0; s1[r] = p1;
      float rs = p0 + p1;
      rs += __shfl_xor(rs, 1, 32);
      rs += __shfl_xor(rs, 2, 32);
      rs += __shfl_xor(rs, 4, 32);
      rs += __shfl_xor(rs, 8, 32);
      lr[r] = lr[r] * alpha[r] + rs;
      mr[r] = mnew;
    }
    #pragma unroll
    for (int nt = 0; nt < 8; ++nt)
      #pragma unroll
      for (int r = 0; r < 8; ++r) accO[nt][r] *= alpha[r];

    #pragma unroll
    for (int r = 0; r < 8; ++r) {
      int m = r + 8 * h2;
      plds[wid][m][lm]      = (__bf16)s0[r];
      plds[wid][m][16 + lm] = (__bf16)s1[r];
    }
    v16bf pf = comb(*(const v8bf*)&plds[wid][lm][h2 * 8],
                    *(const v8bf*)&plds[wid][lm][16 + h2 * 8]);

    #pragma unroll
    for (int nt = 0; nt < 8; ++nt) {
      const __bf16* vp = Vb + ((size_t)bh * ODIM + nt * 16 + lm) * TN + i0;
      v16bf vf = comb(*(const v8bf*)(vp + h2 * 8),
                      *(const v8bf*)(vp + 16 + h2 * 8));
      accO[nt] = wmma_bf16(pf, vf, accO[nt]);
    }
  }

  float inv[8];
  #pragma unroll
  for (int r = 0; r < 8; ++r) inv[r] = 1.0f / lr[r];
  const int bb = bh >> 3;
  const int hh = bh & 7;
  #pragma unroll
  for (int nt = 0; nt < 8; ++nt) {
    #pragma unroll
    for (int r = 0; r < 8; ++r) {
      int e = nt * 16 + lm;
      int t = tb + r + 8 * h2;
      Ot[((size_t)bb * TN + t) * KD + hh * ODIM + e] =
          (__bf16)(accO[nt][r] * inv[r]);
    }
  }
}

// ---------------------------------------------------------------------------
extern "C" void kernel_launch(void* const* d_in, const int* in_sizes, int n_in,
                              void* d_out, int out_size, void* d_ws, size_t ws_size,
                              hipStream_t stream) {
  (void)in_sizes; (void)n_in; (void)out_size; (void)ws_size;
  const float* x  = (const float*)d_in[0];
  const float* Wk = (const float*)d_in[1];
  const float* Wq = (const float*)d_in[2];
  const float* Wv = (const float*)d_in[3];
  const float* Wu = (const float*)d_in[4];
  const float* bu = (const float*)d_in[5];

  // ws (bf16 elems): Qt|Kt|V|Ot|xT (8M each) + Wq/Wk/Wv (1M each) + Wu (128K)
  const size_t SEG = (size_t)4 * HEADS * TN * ODIM;  // 8,388,608
  __bf16* ws  = (__bf16*)d_ws;
  __bf16* Qt  = ws;
  __bf16* Kt  = Qt + SEG;
  __bf16* Vb  = Kt + SEG;
  __bf16* Ot  = Vb + SEG;
  __bf16* xT  = Ot + SEG;
  __bf16* Wqb = xT + SEG;
  __bf16* Wkb = Wqb + (size_t)KD * KD;
  __bf16* Wvb = Wkb + (size_t)KD * KD;
  __bf16* Wub = Wvb + (size_t)KD * KD;

  const float qscale = 0.08838834764831845f;  // 1/sqrt(OD=128)
  dim3 blk(256);

  cvt_bf16<<<dim3(1024), blk, 0, stream>>>(Wq, Wqb, 262144);
  cvt_bf16<<<dim3(1024), blk, 0, stream>>>(Wk, Wkb, 262144);
  cvt_bf16<<<dim3(1024), blk, 0, stream>>>(Wv, Wvb, 262144);
  cvt_bf16<<<dim3(128),  blk, 0, stream>>>(Wu, Wub, 32768);
  transpose_x<<<dim3(64, 32, 4), dim3(32, 8), 0, stream>>>(x, xT);

  gemm_wmma_tdm<1><<<dim3(16, 8, 4), blk, 0, stream>>>(Wqb, xT, Qt, nullptr, qscale);
  gemm_wmma_tdm<1><<<dim3(16, 8, 4), blk, 0, stream>>>(Wkb, xT, Kt, nullptr, 1.0f);
  gemm_wmma_tdm<0><<<dim3(16, 8, 4), blk, 0, stream>>>(Wvb, xT, Vb, nullptr, 1.0f);

  attn_wmma<<<dim3(512), blk, 0, stream>>>(Qt, Kt, Vb, Ot);

  gemm_wmma_tdm<2><<<dim3(16, 1, 4), blk, 0, stream>>>(Wub, Ot, d_out, bu, 1.0f);
}
